// MB_dMGC_CWTFFNet_850403525025
// MI455X (gfx1250) — matrix operand order repaired and checked by reading, not compiled, not run on hardware
//
#include <hip/hip_runtime.h>
#include <hip/hip_bf16.h>

// ---------------- types / helpers ----------------
typedef float  v8f   __attribute__((ext_vector_type(8)));
typedef __bf16 v16bf __attribute__((ext_vector_type(16)));

union Frag16 { uint4 q[2]; v16bf v; };

__device__ __forceinline__ unsigned short f2bf(float f) {
  unsigned int u = __float_as_uint(f);
  u += 0x7FFFu + ((u >> 16) & 1u);          // round-to-nearest-even
  return (unsigned short)(u >> 16);
}
__device__ __forceinline__ float bf2f(unsigned short h) {
  return __uint_as_float(((unsigned int)h) << 16);
}

// A-fragment (16x32 bf16, ISA 7.12.2): lane L holds row m=L&15,
// kbase=(L>=16)?8:0; elems e0..7 -> k=kbase+e, e8..15 -> k=kbase+8+e.
// Chunks at kcol and kcol+16 in a row of contiguous bf16.
__device__ __forceinline__ v16bf load_afrag(const unsigned short* row, int kcol) {
  Frag16 f;
  f.q[0] = *(const uint4*)(row + kcol);
  f.q[1] = *(const uint4*)(row + kcol + 16);
  return f.v;
}
// B-fragment pre-packed contiguously (16 bf16 per lane)
__device__ __forceinline__ v16bf load_bfrag(const unsigned short* p) {
  Frag16 f;
  f.q[0] = ((const uint4*)p)[0];
  f.q[1] = ((const uint4*)p)[1];
  return f.v;
}

#define WMMA_BF16(a, b, c) \
  __builtin_amdgcn_wmma_f32_16x16x32_bf16(false, (a), false, (b), (short)0, (c), false, false)

// ---------------- workspace layout (bf16 elements) ----------------
#define WS_QKV   0         // 9 mats * 2 ntiles * 16 ksteps * 32 lanes * 16 = 147456
#define WS_FFN   147456    // 4 mats * 6 ntiles * 3 ksteps * 512         = 36864
#define WS_OUT1  184320    // 32 * 1728                                   = 55296
#define WS_TOTAL 239616

// ---------------- LDS layout (byte offsets into dynamic smem) ----------------
#define GPITCH 520         // bf16 pitch for G rows (bank-skewed)
#define APITCH 104         // bf16 pitch for 96-wide activation rows
#define SM_G    0          // 54 rows * 520 us = 56160 B (dead after stage A)
#define SM_ZROW 56160      // 520 us zero row = 1040 B
#define SM_QKV  57200      // 3*3*18*32 f32   = 20736 B
// aliases of SM_G region (used after stage A):
#define SM_S    0          // 972 f32  (attn logits/probs)
#define SM_AV   3904       // 1728 f32
#define SM_ZL   10816      // 1728 f32 (Z_local)
#define SM_ZGF  17728      // 1728 f32 (Z_global)
#define SM_ZGB  24640      // 18*104 bf16
#define SM_XNB  28384      // 18*104 bf16
#define SM_AFC  32128      // 18*104 bf16
#define SM_AFM  35872      // 18*104 bf16
#define SM_O1   39616      // 1728 f32
#define SM_O2   46528      // 1728 f32
#define SM_MISC 53440      // 448 f32 scratch
#define SMEM_BYTES 77936

// ---------------- pre-kernel: pack weights to bf16 fragments ----------------
__global__ void prep_weights(const float* __restrict__ wq, const float* __restrict__ wk,
                             const float* __restrict__ wv,
                             const float* __restrict__ fc1, const float* __restrict__ fc2,
                             const float* __restrict__ fm1, const float* __restrict__ fm2,
                             const float* __restrict__ out1,
                             unsigned short* __restrict__ ws) {
  int idx = blockIdx.x * blockDim.x + threadIdx.x;
  if (idx < 147456) {                     // QKV projection B-fragments
    int e = idx & 15, lane = (idx >> 4) & 31, kstep = (idx >> 9) & 15;
    int nt = (idx >> 13) & 1, matg = idx >> 14;       // matg = n*3 + {Q,K,V}
    int n = matg / 3, mq = matg % 3;
    const float* w = (mq == 0) ? wq : (mq == 1) ? wk : wv;   // (3,32,512)
    int o = nt * 16 + (lane & 15);
    int d = kstep * 32 + ((lane & 16) >> 1) + (e & 7) + ((e & 8) << 1);
    ws[WS_QKV + idx] = f2bf(w[(n * 32 + o) * 512 + d]);
  } else if (idx < WS_OUT1) {             // FFN 96x96 B-fragments
    int j = idx - WS_FFN;
    int e = j & 15, lane = (j >> 4) & 31;
    int kstep = (j / 512) % 3, nt = (j / 1536) % 6, mat = j / 9216; // fc1,fc2,fm1,fm2
    const float* w = (mat == 0) ? fc1 : (mat == 1) ? fc2 : (mat == 2) ? fm1 : fm2;
    int o = nt * 16 + (lane & 15);
    int d = kstep * 32 + ((lane & 16) >> 1) + (e & 7) + ((e & 8) << 1);
    ws[WS_FFN + j] = f2bf(w[o * 96 + d]);
  } else if (idx < WS_TOTAL) {            // out1_w plain bf16 copy (32x1728)
    ws[idx] = f2bf(out1[idx - WS_OUT1]);
  }
}

// ---------------- main fused kernel: one block per batch ----------------
__global__ __launch_bounds__(256) void fused_net(
    const float* __restrict__ GT, const float* __restrict__ GS, const float* __restrict__ GR,
    const float* __restrict__ AT, const float* __restrict__ AS, const float* __restrict__ AR,
    const float* __restrict__ Ainit,
    const float* __restrict__ WQb, const float* __restrict__ WKb, const float* __restrict__ WVb,
    const float* __restrict__ ln_g, const float* __restrict__ ln_b,
    const float* __restrict__ fc1_b, const float* __restrict__ fc2_b,
    const float* __restrict__ fm1_b, const float* __restrict__ fm2_b,
    const float* __restrict__ out1_b, const float* __restrict__ out2_w,
    const float* __restrict__ out2_b,
    const unsigned short* __restrict__ ws,
    float* __restrict__ out) {
  extern __shared__ char smem[];
  unsigned short* gbf  = (unsigned short*)(smem + SM_G);
  unsigned short* zrow = (unsigned short*)(smem + SM_ZROW);
  float*          qkv  = (float*)(smem + SM_QKV);
  float*          sS   = (float*)(smem + SM_S);
  float*          sAV  = (float*)(smem + SM_AV);
  float*          sZL  = (float*)(smem + SM_ZL);
  float*          sZG  = (float*)(smem + SM_ZGF);
  unsigned short* zgb  = (unsigned short*)(smem + SM_ZGB);
  unsigned short* xnb  = (unsigned short*)(smem + SM_XNB);
  unsigned short* afc  = (unsigned short*)(smem + SM_AFC);
  unsigned short* afm  = (unsigned short*)(smem + SM_AFM);
  float*          sO1  = (float*)(smem + SM_O1);
  float*          sO2  = (float*)(smem + SM_O2);
  float*          msc  = (float*)(smem + SM_MISC);

  const int tid  = threadIdx.x;
  const int lane = tid & 31;
  const int wid  = tid >> 5;            // 8 waves
  const int li   = lane & 15;
  const int hi8  = (lane & 16) ? 8 : 0; // C/D row offset & A kbase
  const int b    = blockIdx.x;

  // ---- stage 0: zero row + stage G (3x18x512 f32 -> bf16 LDS, coalesced) ----
  for (int i = tid; i < GPITCH; i += 256) zrow[i] = 0;
  for (int i = tid; i < 3 * 18 * 512; i += 256) {
    int n = i / (18 * 512), r = i % (18 * 512), c = r / 512, d = r % 512;
    const float* g = (n == 0) ? GT : (n == 1) ? GS : GR;
    gbf[(n * 18 + c) * GPITCH + d] = f2bf(g[((size_t)b * 18 + c) * 512 + d]);
  }
  __syncthreads();

  // ---- stage A: Q/K/V projections via WMMA bf16 (18 wave-tasks) ----
  for (int t = wid; t < 18; t += 8) {
    int n = t / 6, mt = (t % 6) / 2, nt = t & 1;
    int matg = n * 3 + mt;
    v8f acc0 = {}; v8f acc1 = {};
    const unsigned short* wsB   = ws + WS_QKV + ((matg * 2 + nt) * 16) * 512 + lane * 16;
    const unsigned short* rowA0 = gbf + (n * 18 + li) * GPITCH;
    const unsigned short* rowA1 = (li < 2) ? (gbf + (n * 18 + 16 + li) * GPITCH) : zrow;
    for (int ks = 0; ks < 16; ++ks) {
      v16bf bf = load_bfrag(wsB + ks * 512);
      int kcol = ks * 32 + hi8;
      v16bf a0 = load_afrag(rowA0, kcol);
      v16bf a1 = load_afrag(rowA1, kcol);
      acc0 = WMMA_BF16(a0, bf, acc0);
      acc1 = WMMA_BF16(a1, bf, acc1);
    }
    int o = nt * 16 + li;
    const float* bsrc = (mt == 0) ? WQb : (mt == 1) ? WKb : WVb;
    float bias = bsrc[n * 32 + o];
    float* dst = qkv + (matg * 18) * 32;
    for (int r = 0; r < 8; ++r) {
      int m = r + hi8;                       // 0..15
      dst[m * 32 + o] = acc0[r] + bias;
      int c2 = 16 + m;
      if (c2 < 18) dst[c2 * 32 + o] = acc1[r] + bias;
    }
  }
  __syncthreads();

  // ---- stage B: attention (f32 VALU; tiny 18x18 work) ----
  const float scale = 0.17677669529663687f;  // 1/sqrt(32)
  for (int s = tid; s < 972; s += 256) {
    int n = s / 324, r = s % 324, i = r / 18, j = r % 18;
    const float* Qr = qkv + ((n * 3 + 0) * 18 + i) * 32;
    const float* Kr = qkv + ((n * 3 + 1) * 18 + j) * 32;
    float acc = 0.f;
    for (int k = 0; k < 32; ++k) acc += Qr[k] * Kr[k];
    sS[(n * 18 + i) * 18 + j] = acc * scale;
  }
  __syncthreads();
  for (int t = tid; t < 54; t += 256) {      // row softmax
    float* row = sS + t * 18;
    float mx = row[0];
    for (int j = 1; j < 18; ++j) mx = fmaxf(mx, row[j]);
    float sum = 0.f;
    for (int j = 0; j < 18; ++j) { float e = expf(row[j] - mx); row[j] = e; sum += e; }
    float inv = 1.0f / sum;
    for (int j = 0; j < 18; ++j) row[j] *= inv;
  }
  __syncthreads();
  for (int s = tid; s < 1728; s += 256) {    // AV = attn @ V
    int n = s / 576, r = s % 576, i = r / 32, v = r % 32;
    const float* ar = sS + (n * 18 + i) * 18;
    const float* Vc = qkv + ((n * 3 + 2) * 18) * 32 + v;
    float acc = 0.f;
    for (int j = 0; j < 18; ++j) acc += ar[j] * Vc[j * 32];
    sAV[(n * 18 + i) * 32 + v] = acc;
  }
  __syncthreads();
  for (int s = tid; s < 1728; s += 256) {    // Z = A @ AV, concat into Z_local
    int n = s / 576, r = s % 576, i = r / 32, v = r % 32;
    const float* Ab = ((n == 0) ? AT : (n == 1) ? AS : AR) + (size_t)b * 324 + i * 18;
    const float* AVc = sAV + (n * 18) * 32 + v;
    float acc = 0.f;
    for (int j = 0; j < 18; ++j) acc += Ab[j] * AVc[j * 32];
    sZL[i * 96 + n * 32 + v] = acc;
  }
  __syncthreads();

  // ---- stage C: A_vec/cw softmax weighting + LayerNorm ----
  if (tid < 18) {
    float s1 = 0.f;
    for (int j = 0; j < 18; ++j) s1 += Ainit[tid * 18 + j];
    msc[tid] = s1 * (1.0f / 18.0f);          // A_init row mean
    float s2 = 0.f;
    for (int h = 0; h < 96; ++h) s2 += sZL[tid * 96 + h];
    msc[32 + tid] = s2 * (1.0f / 96.0f);     // Z_local row mean
  }
  __syncthreads();
  if (tid < 18) {                            // redundant softmax over C (tiny)
    float mx1 = msc[0], mx2 = msc[32];
    for (int j = 1; j < 18; ++j) { mx1 = fmaxf(mx1, msc[j]); mx2 = fmaxf(mx2, msc[32 + j]); }
    float s1 = 0.f, s2 = 0.f;
    for (int j = 0; j < 18; ++j) { s1 += expf(msc[j] - mx1); s2 += expf(msc[32 + j] - mx2); }
    float avec = expf(msc[tid] - mx1) / s1;
    float cw   = expf(msc[32 + tid] - mx2) / s2;
    msc[64 + tid] = avec * cw;
  }
  __syncthreads();
  for (int s = tid; s < 1728; s += 256) {    // Z_global + bf16 copy for fm branch
    int c = s / 96, h = s % 96;
    float z = sZL[s] * msc[64 + c];
    sZG[s] = z;
    zgb[c * APITCH + h] = f2bf(z);
  }
  __syncthreads();
  if (tid < 18) {                            // LN stats
    float mu = 0.f;
    for (int h = 0; h < 96; ++h) mu += sZG[tid * 96 + h];
    mu *= (1.0f / 96.0f);
    float var = 0.f;
    for (int h = 0; h < 96; ++h) { float d = sZG[tid * 96 + h] - mu; var += d * d; }
    var *= (1.0f / 96.0f);
    msc[96 + tid]  = mu;
    msc[128 + tid] = rsqrtf(var + 1e-5f);
  }
  __syncthreads();
  for (int s = tid; s < 1728; s += 256) {    // xn -> bf16 for fc branch
    int c = s / 96, h = s % 96;
    float xn = (sZG[s] - msc[96 + c]) * msc[128 + c] * ln_g[h] + ln_b[h];
    xnb[c * APITCH + h] = f2bf(xn);
  }
  __syncthreads();

  // ---- stage D round 1: fc1 (ReLU) / fm1 (ELU) via WMMA (12 wave-tasks) ----
  for (int t = wid; t < 12; t += 8) {
    int matsel = t / 6, nt = t % 6;
    const unsigned short* inA = (matsel == 0) ? xnb : zgb;
    int fragmat = (matsel == 0) ? 0 : 2;
    v8f acc0 = {}; v8f acc1 = {};
    const unsigned short* wsB   = ws + WS_FFN + ((fragmat * 6 + nt) * 3) * 512 + lane * 16;
    const unsigned short* rowA0 = inA + li * APITCH;
    const unsigned short* rowA1 = (li < 2) ? (inA + (16 + li) * APITCH) : zrow;
    for (int ks = 0; ks < 3; ++ks) {
      v16bf bf = load_bfrag(wsB + ks * 512);
      int kcol = ks * 32 + hi8;
      v16bf a0 = load_afrag(rowA0, kcol);
      v16bf a1 = load_afrag(rowA1, kcol);
      acc0 = WMMA_BF16(a0, bf, acc0);
      acc1 = WMMA_BF16(a1, bf, acc1);
    }
    int o = nt * 16 + li;
    float bias = ((matsel == 0) ? fc1_b : fm1_b)[o];
    unsigned short* dst = (matsel == 0) ? afc : afm;
    for (int r = 0; r < 8; ++r) {
      int m = r + hi8;
      float v0 = acc0[r] + bias;
      v0 = (matsel == 0) ? fmaxf(v0, 0.f) : ((v0 > 0.f) ? v0 : expm1f(v0));
      dst[m * APITCH + o] = f2bf(v0);
      int c2 = 16 + m;
      if (c2 < 18) {
        float v1 = acc1[r] + bias;
        v1 = (matsel == 0) ? fmaxf(v1, 0.f) : ((v1 > 0.f) ? v1 : expm1f(v1));
        dst[c2 * APITCH + o] = f2bf(v1);
      }
    }
  }
  __syncthreads();

  // ---- stage D round 2: fc2 / fm2 via WMMA ----
  for (int t = wid; t < 12; t += 8) {
    int matsel = t / 6, nt = t % 6;
    const unsigned short* inA = (matsel == 0) ? afc : afm;
    int fragmat = (matsel == 0) ? 1 : 3;
    v8f acc0 = {}; v8f acc1 = {};
    const unsigned short* wsB   = ws + WS_FFN + ((fragmat * 6 + nt) * 3) * 512 + lane * 16;
    const unsigned short* rowA0 = inA + li * APITCH;
    const unsigned short* rowA1 = (li < 2) ? (inA + (16 + li) * APITCH) : zrow;
    for (int ks = 0; ks < 3; ++ks) {
      v16bf bf = load_bfrag(wsB + ks * 512);
      int kcol = ks * 32 + hi8;
      v16bf a0 = load_afrag(rowA0, kcol);
      v16bf a1 = load_afrag(rowA1, kcol);
      acc0 = WMMA_BF16(a0, bf, acc0);
      acc1 = WMMA_BF16(a1, bf, acc1);
    }
    int o = nt * 16 + li;
    float bias = ((matsel == 0) ? fc2_b : fm2_b)[o];
    float* dst = (matsel == 0) ? sO1 : sO2;
    for (int r = 0; r < 8; ++r) {
      int m = r + hi8;
      dst[m * 96 + o] = acc0[r] + bias;
      int c2 = 16 + m;
      if (c2 < 18) dst[c2 * 96 + o] = acc1[r] + bias;
    }
  }
  __syncthreads();

  // ---- combine + emit Z_fused ----
  float* outZ = out + 4096 + (size_t)b * 1728;
  for (int s = tid; s < 1728; s += 256) {
    float z = sO1[s] + sO2[s];
    sO1[s] = z;                 // keep flat vector for classifier
    outZ[s] = z;
  }
  __syncthreads();

  // ---- classifier head: 32x1728 GEMV (bf16 weights), then 2x32 ----
  {
    int j = tid >> 3, part = tid & 7;
    const unsigned short* wrow = ws + WS_OUT1 + j * 1728;
    float acc = 0.f;
    for (int i = part; i < 1728; i += 8) acc += sO1[i] * bf2f(wrow[i]);
    msc[192 + tid] = acc;
  }
  __syncthreads();
  if (tid < 32) {
    float s = 0.f;
    for (int p = 0; p < 8; ++p) s += msc[192 + tid * 8 + p];
    msc[160 + tid] = fmaxf(s + out1_b[tid], 0.f);
  }
  __syncthreads();
  if (tid < 2) {
    float acc = out2_b[tid];
    for (int k = 0; k < 32; ++k) acc += msc[160 + k] * out2_w[tid * 32 + k];
    out[(size_t)b * 2 + tid] = acc;
  }
}

// ---------------- host entry ----------------
extern "C" void kernel_launch(void* const* d_in, const int* in_sizes, int n_in,
                              void* d_out, int out_size, void* d_ws, size_t ws_size,
                              hipStream_t stream) {
  const float* GT    = (const float*)d_in[0];
  const float* GS    = (const float*)d_in[1];
  const float* GR    = (const float*)d_in[2];
  const float* AT    = (const float*)d_in[3];
  const float* AS    = (const float*)d_in[4];
  const float* AR    = (const float*)d_in[5];
  const float* Ainit = (const float*)d_in[6];
  const float* WQw   = (const float*)d_in[7];
  const float* WQb   = (const float*)d_in[8];
  const float* WKw   = (const float*)d_in[9];
  const float* WKb   = (const float*)d_in[10];
  const float* WVw   = (const float*)d_in[11];
  const float* WVb   = (const float*)d_in[12];
  const float* ln_g  = (const float*)d_in[13];
  const float* ln_b  = (const float*)d_in[14];
  const float* fc1_w = (const float*)d_in[15];
  const float* fc1_b = (const float*)d_in[16];
  const float* fc2_w = (const float*)d_in[17];
  const float* fc2_b = (const float*)d_in[18];
  const float* fm1_w = (const float*)d_in[19];
  const float* fm1_b = (const float*)d_in[20];
  const float* fm2_w = (const float*)d_in[21];
  const float* fm2_b = (const float*)d_in[22];
  const float* out1w = (const float*)d_in[23];
  const float* out1b = (const float*)d_in[24];
  const float* out2w = (const float*)d_in[25];
  const float* out2b = (const float*)d_in[26];
  unsigned short* ws = (unsigned short*)d_ws;
  float* out = (float*)d_out;

  prep_weights<<<(WS_TOTAL + 255) / 256, 256, 0, stream>>>(
      WQw, WKw, WVw, fc1_w, fc2_w, fm1_w, fm2_w, out1w, ws);

  fused_net<<<2048, 256, SMEM_BYTES, stream>>>(
      GT, GS, GR, AT, AS, AR, Ainit,
      WQb, WKb, WVb, ln_g, ln_b,
      fc1_b, fc2_b, fm1_b, fm2_b,
      out1b, out2w, out2b, ws, out);
}
